// CrossAttention_9174050144362
// MI455X (gfx1250) — matrix operand order repaired and checked by reading, not compiled, not run on hardware
//
#include <hip/hip_runtime.h>
#include <hip/hip_bf16.h>
#include <math.h>

#define BB 4
#define CC 256
#define CHD 128
#define HW 4096

typedef __attribute__((ext_vector_type(16))) _Float16 v16h;
typedef __attribute__((ext_vector_type(8)))  float    v8f;

// Load a 16-half WMMA fragment as two 16-byte chunks (global or LDS).
static __device__ __forceinline__ v16h ld_frag2(const _Float16* p0, const _Float16* p1) {
    union { v16h v; float4 f[2]; } u;
    u.f[0] = *(const float4*)p0;
    u.f[1] = *(const float4*)p1;
    return u.v;
}

// ---------------------------------------------------------------------------
// Transpose+convert: X [B][256][HW] f32  ->  Xt [B][HW][256] f16
// 32x32 LDS tile, 256 threads/block.
// ---------------------------------------------------------------------------
__global__ __launch_bounds__(256)
void transpose_f16_kernel(const float* __restrict__ X, _Float16* __restrict__ Xt) {
    __shared__ float sT[32][33];
    const int m0 = blockIdx.x * 32;
    const int c0 = blockIdx.y * 32;
    const int b  = blockIdx.z;
    const int tx = threadIdx.x & 31;
    const int ty = threadIdx.x >> 5;            // 0..7

    const float* Xb = X + (size_t)b * CC * HW;
#pragma unroll
    for (int i = 0; i < 4; ++i) {
        const int r = ty + i * 8;
        sT[r][tx] = Xb[(size_t)(c0 + r) * HW + m0 + tx];
    }
    __syncthreads();
    _Float16* Ob = Xt + (size_t)b * HW * CC;
#pragma unroll
    for (int i = 0; i < 4; ++i) {
        const int r = ty + i * 8;
        Ob[(size_t)(m0 + r) * CC + c0 + tx] = (_Float16)sT[tx][r];
    }
}

// f32 -> f16 elementwise (for weights)
__global__ void cvt_f16_kernel(const float* __restrict__ src, _Float16* __restrict__ dst, int n) {
    const int i = blockIdx.x * 256 + threadIdx.x;
    if (i < n) dst[i] = (_Float16)src[i];
}

// ---------------------------------------------------------------------------
// WMMA projection: Out = Xt . W^T + b.  One wave per 16m x 16c tile, K = 256.
// Xt: [B][HW][256] f16 (A frags), Wh: [Cout][256] f16 (B frags, natural layout)
// transpose_out=1 -> Out[b][m][Cout] ; else Out[b][Cout][m]
// ---------------------------------------------------------------------------
__global__ __launch_bounds__(128)
void proj_wmma_kernel(const _Float16* __restrict__ Xt, const _Float16* __restrict__ Wh,
                      const float* __restrict__ bias, _Float16* __restrict__ Outp,
                      int Cout, int transpose_out) {
    const int lane = threadIdx.x & 31;
    const int wid  = threadIdx.x >> 5;
    const int lo   = lane & 15;
    const int hi   = (lane >> 4) & 1;
    const int c0   = (blockIdx.x * 4 + wid) * 16;
    const int m0   = blockIdx.y * 16;
    const int b    = blockIdx.z;

    const _Float16* xrow = Xt + ((size_t)b * HW + m0 + lo) * CC;
    const _Float16* wrow = Wh + (size_t)(c0 + lo) * CC;

    v8f acc = {};
#pragma unroll
    for (int kk = 0; kk < 8; ++kk) {
        const _Float16* ap = xrow + kk * 32 + hi * 8;
        v16h af = ld_frag2(ap, ap + 16);
        const _Float16* bp = wrow + kk * 32 + hi * 8;
        v16h bf = ld_frag2(bp, bp + 16);
        acc = __builtin_amdgcn_wmma_f32_16x16x32_f16(false, af, false, bf,
                                                     (short)0, acc, false, false);
    }
    const float bv = bias[c0 + lo];
    if (transpose_out) {
#pragma unroll
        for (int r = 0; r < 8; ++r) {
            const int m = m0 + r + 8 * hi;
            Outp[((size_t)b * HW + m) * (size_t)Cout + (c0 + lo)] = (_Float16)(acc[r] + bv);
        }
    } else {
#pragma unroll
        for (int r = 0; r < 8; ++r) {
            const int n = m0 + r + 8 * hi;
            Outp[((size_t)b * Cout + (c0 + lo)) * (size_t)HW + n] = (_Float16)(acc[r] + bv);
        }
    }
}

// ---------------------------------------------------------------------------
// Flash-attention. Block = 4 waves = 64 query rows; K/V tiles staged in LDS
// once per block per 32-column step (4x less L2 traffic than per-wave loads).
// ---------------------------------------------------------------------------
__global__ __launch_bounds__(128, 1)
void attn_kernel(const _Float16* __restrict__ Qt, const _Float16* __restrict__ Kt,
                 const _Float16* __restrict__ Vm, const float* __restrict__ Ares,
                 float* __restrict__ Out) {
    __shared__ _Float16 sK[32][CHD];            // 8 KB   [n_local][c]
    __shared__ _Float16 sV[CC][32];             // 16 KB  [c][n_local]
    __shared__ _Float16 sP[4][16][32];          // 4 KB   per-wave P staging

    const int tid  = threadIdx.x;
    const int lane = tid & 31;
    const int wid  = tid >> 5;
    const int lo   = lane & 15;
    const int hi   = (lane >> 4) & 1;
    const int b    = blockIdx.x >> 6;           // 64 m-blocks per batch
    const int m0   = (blockIdx.x & 63) * 64 + wid * 16;

    const _Float16* Qb = Qt + (size_t)b * HW * CHD;
    const _Float16* Kb = Kt + (size_t)b * HW * CHD;
    const _Float16* Vb = Vm + (size_t)b * CC * HW;

    v16h qa[4];
    {
        const _Float16* qrow = Qb + (size_t)(m0 + lo) * CHD;
#pragma unroll
        for (int kk = 0; kk < 4; ++kk) {
            const _Float16* p0 = qrow + kk * 32 + hi * 8;
            qa[kk] = ld_frag2(p0, p0 + 16);
        }
    }

    const v8f vzero = {};
    v8f Oacc[16];
#pragma unroll
    for (int t = 0; t < 16; ++t) Oacc[t] = vzero;

    float rowm[8], rowl[8];
#pragma unroll
    for (int r = 0; r < 8; ++r) { rowm[r] = -INFINITY; rowl[r] = 0.f; }

    for (int n0 = 0; n0 < HW; n0 += 32) {
        // ---- cooperative K/V tile staging: all global loads issued first ----
        float4 tk[4], tv[8];
#pragma unroll
        for (int i = 0; i < 4; ++i) {           // K tile: 32 rows x 256B
            const int ch = i * 128 + tid;
            const int r  = ch >> 4, c8 = (ch & 15) * 8;
            tk[i] = *(const float4*)(Kb + (size_t)(n0 + r) * CHD + c8);
        }
#pragma unroll
        for (int i = 0; i < 8; ++i) {           // V tile: 256 rows x 64B
            const int ch = i * 128 + tid;
            const int r  = ch >> 2, c8 = (ch & 3) * 8;
            tv[i] = *(const float4*)(Vb + (size_t)r * HW + n0 + c8);
        }
#pragma unroll
        for (int i = 0; i < 4; ++i) {
            const int ch = i * 128 + tid;
            *(float4*)(&sK[ch >> 4][(ch & 15) * 8]) = tk[i];
        }
#pragma unroll
        for (int i = 0; i < 8; ++i) {
            const int ch = i * 128 + tid;
            *(float4*)(&sV[ch >> 2][(ch & 3) * 8]) = tv[i];
        }
        __syncthreads();

        // prefetch next step's tiles into cache while this step computes
        if (n0 + 32 < HW) {
            __builtin_prefetch(Kb + (size_t)(n0 + 32 + lane) * CHD + (tid >> 5) * 32, 0, 0);
            __builtin_prefetch(Vb + (size_t)(tid * 2) * HW + n0 + 32, 0, 0);
        }

        // ---- S = Q^T K : 8 pipelined WMMAs (2 n-subtiles x 4 c-chunks) ----
        v8f s0 = vzero, s1 = vzero;
        {
            const _Float16* p = &sK[lo][hi * 8];
            v16h cur = ld_frag2(p, p + 16);
#pragma unroll
            for (int i = 0; i < 8; ++i) {
                v16h nxt = cur;
                if (i < 7) {
                    const int ni = i + 1;
                    const _Float16* q = &sK[(ni & 1) * 16 + lo][(ni >> 1) * 32 + hi * 8];
                    nxt = ld_frag2(q, q + 16);
                }
                if ((i & 1) == 0)
                    s0 = __builtin_amdgcn_wmma_f32_16x16x32_f16(false, qa[i >> 1], false, cur,
                                                                (short)0, s0, false, false);
                else
                    s1 = __builtin_amdgcn_wmma_f32_16x16x32_f16(false, qa[i >> 1], false, cur,
                                                                (short)0, s1, false, false);
                cur = nxt;
            }
        }

        // ---- online softmax (16-lane reductions keep hi/lo halves apart) ----
#pragma unroll
        for (int r = 0; r < 8; ++r) {
            float v0 = s0[r], v1 = s1[r];
            float mx = fmaxf(v0, v1);
#pragma unroll
            for (int d = 1; d < 16; d <<= 1) mx = fmaxf(mx, __shfl_xor(mx, d, 32));
            float mnew  = fmaxf(rowm[r], mx);
            float alpha = __expf(rowm[r] - mnew);
            float p0 = __expf(v0 - mnew);
            float p1 = __expf(v1 - mnew);
            float ps = p0 + p1;
#pragma unroll
            for (int d = 1; d < 16; d <<= 1) ps += __shfl_xor(ps, d, 32);
            rowl[r] = rowl[r] * alpha + ps;
            rowm[r] = mnew;
#pragma unroll
            for (int t = 0; t < 16; ++t) Oacc[t][r] *= alpha;   // rescale O
            sP[wid][r + 8 * hi][lo]      = (_Float16)p0;        // C layout -> [m][n]
            sP[wid][r + 8 * hi][16 + lo] = (_Float16)p1;
        }
        asm volatile("s_wait_dscnt 0" ::: "memory");

        // reload P as a 16x32 A fragment
        v16h pa;
        {
            const _Float16* prow = &sP[wid][lo][0];
            const _Float16* p0   = prow + hi * 8;
            pa = ld_frag2(p0, p0 + 16);
        }

        // ---- O += P @ V^T : 16 pipelined WMMAs over channel tiles ----
        {
            const _Float16* p = &sV[lo][hi * 8];
            v16h cur = ld_frag2(p, p + 16);
#pragma unroll
            for (int t = 0; t < 16; ++t) {
                v16h nxt = cur;
                if (t < 15) {
                    const _Float16* q = &sV[(t + 1) * 16 + lo][hi * 8];
                    nxt = ld_frag2(q, q + 16);
                }
                Oacc[t] = __builtin_amdgcn_wmma_f32_16x16x32_f16(false, pa, false, cur,
                                                                 (short)0, Oacc[t], false, false);
                cur = nxt;
            }
        }
        __syncthreads();        // protect sK/sV before next stage
    }

    // ---- epilogue: normalize, add residual, store f32 ----
#pragma unroll
    for (int t = 0; t < 16; ++t) {
#pragma unroll
        for (int r = 0; r < 8; ++r) {
            const int c = t * 16 + lo;
            const int m = m0 + r + 8 * hi;
            const size_t idx = ((size_t)b * CC + c) * HW + m;
            Out[idx] = Oacc[t][r] / rowl[r] + Ares[idx];
        }
    }
}

extern "C" void kernel_launch(void* const* d_in, const int* in_sizes, int n_in,
                              void* d_out, int out_size, void* d_ws, size_t ws_size,
                              hipStream_t stream) {
    const float* a  = (const float*)d_in[0];
    const float* p  = (const float*)d_in[1];
    const float* Wq = (const float*)d_in[2];
    const float* bq = (const float*)d_in[3];
    const float* Wk = (const float*)d_in[4];
    const float* bk = (const float*)d_in[5];
    const float* Wv = (const float*)d_in[6];
    const float* bv = (const float*)d_in[7];
    float* out = (float*)d_out;

    // workspace layout (f16 elements)
    _Float16* At  = (_Float16*)d_ws;                      // [B][HW][256]  8 MB
    _Float16* Pt  = At  + (size_t)BB * HW * CC;           // [B][HW][256]  8 MB
    _Float16* Qt  = Pt  + (size_t)BB * HW * CC;           // [B][HW][128]  4 MB
    _Float16* Kt  = Qt  + (size_t)BB * HW * CHD;          // [B][HW][128]  4 MB
    _Float16* Vm  = Kt  + (size_t)BB * HW * CHD;          // [B][256][HW]  8 MB
    _Float16* Whq = Vm  + (size_t)BB * CC * HW;           // [128][256]
    _Float16* Whk = Whq + (size_t)CHD * CC;
    _Float16* Whv = Whk + (size_t)CHD * CC;               // [256][256]

    // 1) transpose+convert activations
    transpose_f16_kernel<<<dim3(HW / 32, CC / 32, BB), dim3(256), 0, stream>>>(a, At);
    transpose_f16_kernel<<<dim3(HW / 32, CC / 32, BB), dim3(256), 0, stream>>>(p, Pt);

    // 2) convert weights to f16
    cvt_f16_kernel<<<dim3((CHD * CC + 255) / 256), dim3(256), 0, stream>>>(Wq, Whq, CHD * CC);
    cvt_f16_kernel<<<dim3((CHD * CC + 255) / 256), dim3(256), 0, stream>>>(Wk, Whk, CHD * CC);
    cvt_f16_kernel<<<dim3((CC * CC + 255) / 256), dim3(256), 0, stream>>>(Wv, Whv, CC * CC);

    // 3) WMMA projections
    proj_wmma_kernel<<<dim3(CHD / 64, HW / 16, BB), dim3(128), 0, stream>>>(At, Whq, bq, Qt, CHD, 1);
    proj_wmma_kernel<<<dim3(CHD / 64, HW / 16, BB), dim3(128), 0, stream>>>(Pt, Whk, bk, Kt, CHD, 1);
    proj_wmma_kernel<<<dim3(CC  / 64, HW / 16, BB), dim3(128), 0, stream>>>(Pt, Whv, bv, Vm, CC, 0);

    // 4) flash attention + residual
    attn_kernel<<<dim3(BB * (HW / 64)), dim3(128), 0, stream>>>(Qt, Kt, Vm, a, out);
}
